// multi_head_attention_65111704207723
// MI455X (gfx1250) — matrix operand order repaired and checked by reading, not compiled, not run on hardware
//
#include <hip/hip_runtime.h>
#include <hip/hip_bf16.h>

typedef __bf16 bf16_t;
typedef __attribute__((ext_vector_type(8)))  __bf16 v8bf;
typedef __attribute__((ext_vector_type(16))) __bf16 v16bf;
typedef __attribute__((ext_vector_type(8)))  float  v8f;
typedef int v4i_ __attribute__((vector_size(16)));   // matches builtin prototype

#define NB   4
#define SEQ  1024
#define DM   1024
#define NH   16
#define DK   64

#define WMMA_BF16(a, b, c) \
  __builtin_amdgcn_wmma_f32_16x16x32_bf16(false, (a), false, (b), (short)0, (c), false, false)

#define HAVE_ASYNC_LDS (__has_builtin(__builtin_amdgcn_global_load_async_to_lds_b128) && \
                        __has_builtin(__builtin_amdgcn_s_wait_asynccnt))
#define HAVE_DPP __has_builtin(__builtin_amdgcn_update_dpp)

static __device__ __forceinline__ v16bf bfcat(v8bf lo, v8bf hi) {
  return __builtin_shufflevector(lo, hi, 0,1,2,3,4,5,6,7,8,9,10,11,12,13,14,15);
}

// A-operand (16x32 bf16, MxK) from a row-major bf16 source.
// lane l: row = l%16, K elems {ko..ko+7, ko+16..ko+23}, ko = 8*(l/16)
static __device__ __forceinline__ v16bf load_a_bf(const bf16_t* base, int stride) {
  int l = threadIdx.x & 31;
  const bf16_t* p = base + (size_t)(l & 15) * stride + ((l >> 4) << 3);
  v8bf lo = *(const v8bf*)p;
  v8bf hi = *(const v8bf*)(p + 16);
  return bfcat(lo, hi);
}

// B-operand (32x16 bf16, KxN) from a row-major source holding B-TRANSPOSED
// (rows = N, contiguous axis = K). lane l: row n = l%16, K elems {ko..ko+7, ko+16..ko+23}, ko = 8*(l/16)
static __device__ __forceinline__ v16bf load_b_bf(const bf16_t* base, int stride) {
  int l = threadIdx.x & 31;
  const bf16_t* p = base + (size_t)(l & 15) * stride + ((l >> 4) << 3);
  v8bf lo = *(const v8bf*)p;
  v8bf hi = *(const v8bf*)(p + 16);
  return bfcat(lo, hi);
}

// A-operand built from an fp32 row-major source, converting to bf16 on the fly.
static __device__ __forceinline__ v16bf load_a_f32(const float* base, int stride) {
  int l = threadIdx.x & 31;
  const float* p = base + (size_t)(l & 15) * stride + ((l >> 4) << 3);
  v16bf r;
#pragma unroll
  for (int i = 0; i < 8; ++i) {
    r[i]     = (bf16_t)p[i];
    r[i + 8] = (bf16_t)p[i + 16];
  }
  return r;
}

// ---- 16-lane-row reductions (DPP16, zero LDS traffic) ----
#if HAVE_DPP
template <int CTRL>
static __device__ __forceinline__ float dpp_xfer(float x) {
  return __int_as_float(__builtin_amdgcn_update_dpp(
      0, __float_as_int(x), CTRL, 0xf, 0xf, true));
}
static __device__ __forceinline__ float red16_max(float x) {
  x = fmaxf(x, dpp_xfer<0xB1>(x));    // quad_perm(1,0,3,2)  : xor 1
  x = fmaxf(x, dpp_xfer<0x4E>(x));    // quad_perm(2,3,0,1)  : xor 2
  x = fmaxf(x, dpp_xfer<0x141>(x));   // row_half_mirror     : xor 7
  x = fmaxf(x, dpp_xfer<0x140>(x));   // row_mirror          : xor 15
  return x;
}
static __device__ __forceinline__ float red16_sum(float x) {
  x += dpp_xfer<0xB1>(x);
  x += dpp_xfer<0x4E>(x);
  x += dpp_xfer<0x141>(x);
  x += dpp_xfer<0x140>(x);
  return x;
}
#else
static __device__ __forceinline__ float red16_max(float x) {
  x = fmaxf(x, __shfl_xor(x, 1));
  x = fmaxf(x, __shfl_xor(x, 2));
  x = fmaxf(x, __shfl_xor(x, 4));
  x = fmaxf(x, __shfl_xor(x, 8));
  return x;
}
static __device__ __forceinline__ float red16_sum(float x) {
  x += __shfl_xor(x, 1);
  x += __shfl_xor(x, 2);
  x += __shfl_xor(x, 4);
  x += __shfl_xor(x, 8);
  return x;
}
#endif

// C = A[M,K] @ W[K,N] + bias, A fp32 or bf16, W fp32 (staged->bf16 in LDS).
// OUT_MODE 0: bf16 row-major [M,N]
// OUT_MODE 1: bf16 per-head transposed Vt[b,h,d,seq]
// OUT_MODE 2: fp32 row-major [M,N]
template <int OUT_MODE, bool A_BF16>
__global__ __launch_bounds__(128)
void gemm_bias_kernel(const void* __restrict__ Av, const float* __restrict__ W,
                      const float* __restrict__ bias, void* __restrict__ Outv,
                      int K, int N) {
  __shared__ bf16_t wt[64][40];           // [n in tile][k chunk(32) + pad(8)]
  const int tileN = blockIdx.x * 64;
  const int tileM = blockIdx.y * 64;
  const int wid   = threadIdx.x >> 5;
  const int lane  = threadIdx.x & 31;
  const int rowBase = tileM + wid * 16;

  v8f acc[4];
#pragma unroll
  for (int i = 0; i < 4; ++i) acc[i] = {};

  for (int k0 = 0; k0 < K; k0 += 32) {
    // cooperative: W[k0..k0+32) x [tileN..tileN+64) -> LDS transposed bf16
#pragma unroll
    for (int i = 0; i < 16; ++i) {
      int idx = threadIdx.x + i * 128;
      int kk = idx >> 6;        // 0..31
      int nn = idx & 63;        // 0..63
      wt[nn][kk] = (bf16_t)W[(size_t)(k0 + kk) * N + tileN + nn];
    }
    __syncthreads();

    v16bf a;
    if (A_BF16) a = load_a_bf((const bf16_t*)Av + (size_t)rowBase * K + k0, K);
    else        a = load_a_f32((const float*)Av + (size_t)rowBase * K + k0, K);

#pragma unroll
    for (int nt = 0; nt < 4; ++nt) {
      v16bf b = load_b_bf(&wt[nt * 16][0], 40);
      acc[nt] = WMMA_BF16(a, b, acc[nt]);
    }
    __syncthreads();
  }

  const int colLo = lane & 15;
  const int rOff  = (lane >> 4) * 8;
#pragma unroll
  for (int nt = 0; nt < 4; ++nt) {
#pragma unroll
    for (int v = 0; v < 8; ++v) {
      int row = rowBase + rOff + v;
      int col = tileN + nt * 16 + colLo;
      float val = acc[nt][v] + bias[col];
      if (OUT_MODE == 0) {
        ((bf16_t*)Outv)[(size_t)row * N + col] = (bf16_t)val;
      } else if (OUT_MODE == 1) {
        int b = row >> 10, seq = row & 1023;
        int h = col >> 6,  d   = col & 63;
        ((bf16_t*)Outv)[(((size_t)(b * NH + h)) * DK + d) * SEQ + seq] = (bf16_t)val;
      } else {
        ((float*)Outv)[(size_t)row * N + col] = val;
      }
    }
  }
}

#if HAVE_ASYNC_LDS
// async-copy one 16x32 fp32 bias tile (2 KB) into a per-wave LDS buffer:
// 128 x 16B segments, 4 per lane.
static __device__ __forceinline__ void issue_bias_async(const float* gbase,
                                                        float* lbuf, int lane) {
#pragma unroll
  for (int i = 0; i < 4; ++i) {
    int seg = lane + 32 * i;           // 0..127
    int row = seg >> 3;                // 16 rows
    int c16 = seg & 7;                 // 8 x 16B per 128B row
    const float* g = gbase + (size_t)row * SEQ + c16 * 4;
    float* l = lbuf + row * 32 + c16 * 4;
    __builtin_amdgcn_global_load_async_to_lds_b128(
        (__attribute__((address_space(1))) v4i_*)g,
        (__attribute__((address_space(3))) v4i_*)l, 0, 0);
  }
}
#endif

// Flash attention: one wave per 16-query tile per (b,h). 32 keys per iteration.
__global__ __launch_bounds__(128)
void attn_kernel(const bf16_t* __restrict__ Q, const bf16_t* __restrict__ Kb,
                 const bf16_t* __restrict__ Vt, const float* __restrict__ bias,
                 bf16_t* __restrict__ Ctx) {
  __shared__ bf16_t plds[4][16][40];      // per-wave P tile (16 q x 32 k + pad)
#if HAVE_ASYNC_LDS
  __shared__ float blds[4][2][16 * 32];   // per-wave double-buffered bias tile
#endif
  const int wid  = threadIdx.x >> 5;
  const int lane = threadIdx.x & 31;
  const int wg = blockIdx.x * 4 + wid;    // 0..4095
  const int qt = wg & 63;
  const int h  = (wg >> 6) & 15;
  const int b  = wg >> 10;
  const int qbase = qt * 16;

  const bf16_t* qptr = Q  + ((size_t)(b * SEQ + qbase)) * DM + h * DK;
  const bf16_t* kpt0 = Kb + ((size_t)(b * SEQ)) * DM + h * DK;
  const bf16_t* vtp  = Vt + ((size_t)(b * NH + h)) * DK * SEQ;
  const float*  bpt  = bias + (((size_t)(b * NH + h)) * SEQ + qbase) * SEQ;

  const v16bf qa0 = load_a_bf(qptr, DM);        // d = 0..31
  const v16bf qa1 = load_a_bf(qptr + 32, DM);   // d = 32..63

  v8f o[4];
#pragma unroll
  for (int i = 0; i < 4; ++i) o[i] = {};
  float m[8], lsum[8];
#pragma unroll
  for (int v = 0; v < 8; ++v) { m[v] = -3.0e38f; lsum[v] = 0.f; }

  const int colLo = lane & 15;
  const int rOff  = (lane >> 4) * 8;

#if HAVE_ASYNC_LDS
  issue_bias_async(bpt, &blds[wid][0][0], lane);   // prefetch tile for kbq = 0
#endif

  for (int kbq = 0; kbq < SEQ; kbq += 32) {
    const bf16_t* kp = kpt0 + (size_t)kbq * DM;
    v8f s0 = {}, s1 = {};
    {
      v16bf b0 = load_b_bf(kp, DM);                 // keys kbq..+15, d 0..31
      s0 = WMMA_BF16(qa0, b0, s0);
      v16bf b1 = load_b_bf(kp + 32, DM);            // d 32..63
      s0 = WMMA_BF16(qa1, b1, s0);
      v16bf b2 = load_b_bf(kp + 16 * DM, DM);       // keys kbq+16..+31
      s1 = WMMA_BF16(qa0, b2, s1);
      v16bf b3 = load_b_bf(kp + 16 * DM + 32, DM);
      s1 = WMMA_BF16(qa1, b3, s1);
    }

    // scale + additive bias (dominant HBM stream)
#if HAVE_ASYNC_LDS
    {
      const int bufi = (kbq >> 5) & 1;
      if (kbq + 32 < SEQ) {
        issue_bias_async(bpt + (kbq + 32), &blds[wid][bufi ^ 1][0], lane);
        __builtin_amdgcn_s_wait_asynccnt(4);   // current tile done, next in flight
      } else {
        __builtin_amdgcn_s_wait_asynccnt(0);
      }
      const float* lb = &blds[wid][bufi][0];
#pragma unroll
      for (int v = 0; v < 8; ++v) {
        float bi0 = lb[(rOff + v) * 32 + colLo];
        float bi1 = lb[(rOff + v) * 32 + 16 + colLo];
        s0[v] = s0[v] * 0.125f + bi0;
        s1[v] = s1[v] * 0.125f + bi1;
      }
    }
#else
    {
      const float* bp = bpt + kbq + colLo;
#pragma unroll
      for (int v = 0; v < 8; ++v) {
        float bi0 = bp[(size_t)(rOff + v) * SEQ];
        float bi1 = bp[(size_t)(rOff + v) * SEQ + 16];
        s0[v] = s0[v] * 0.125f + bi0;
        s1[v] = s1[v] * 0.125f + bi1;
      }
    }
#endif

    // online softmax (rows live in 16-lane groups; DPP16 reductions)
#pragma unroll
    for (int v = 0; v < 8; ++v) {
      float rm = red16_max(fmaxf(s0[v], s1[v]));
      float mn = fmaxf(m[v], rm);
      float alpha = __expf(m[v] - mn);
      m[v] = mn;
      float p0 = __expf(s0[v] - mn);
      float p1 = __expf(s1[v] - mn);
      float rs = red16_sum(p0 + p1);
      lsum[v] = lsum[v] * alpha + rs;
#pragma unroll
      for (int dt = 0; dt < 4; ++dt) o[dt][v] *= alpha;
      plds[wid][rOff + v][colLo]      = (bf16_t)p0;   // D-layout -> LDS transpose
      plds[wid][rOff + v][16 + colLo] = (bf16_t)p1;
    }
    v16bf pa = load_a_bf(&plds[wid][0][0], 40);       // P in A-layout
#pragma unroll
    for (int dt = 0; dt < 4; ++dt) {
      v16bf vb = load_b_bf(vtp + (size_t)(dt * 16) * SEQ + kbq, SEQ);
      o[dt] = WMMA_BF16(pa, vb, o[dt]);
    }
  }
  // normalize, store ctx as bf16 [b*SEQ+q][h*64+d]
#pragma unroll
  for (int v = 0; v < 8; ++v) {
    float inv = 1.0f / lsum[v];
    int row = qbase + rOff + v;
#pragma unroll
    for (int dt = 0; dt < 4; ++dt) {
      Ctx[((size_t)(b * SEQ + row)) * DM + h * DK + dt * 16 + colLo] =
          (bf16_t)(o[dt][v] * inv);
    }
  }
}

extern "C" void kernel_launch(void* const* d_in, const int* in_sizes, int n_in,
                              void* d_out, int out_size, void* d_ws, size_t ws_size,
                              hipStream_t stream) {
  const float* queries = (const float*)d_in[0];
  const float* keys    = (const float*)d_in[1];
  const float* values  = (const float*)d_in[2];
  const float* attnb   = (const float*)d_in[3];
  const float* Wq = (const float*)d_in[4];  const float* bq = (const float*)d_in[5];
  const float* Wk = (const float*)d_in[6];  const float* bk = (const float*)d_in[7];
  const float* Wv = (const float*)d_in[8];  const float* bv = (const float*)d_in[9];
  const float* Wo = (const float*)d_in[10]; const float* bo = (const float*)d_in[11];

  const int M = NB * SEQ;                       // 4096
  char* ws = (char*)d_ws;
  bf16_t* Qbf = (bf16_t*)(ws);                               // 8 MB
  bf16_t* Kbf = (bf16_t*)(ws + (size_t)8  * 1024 * 1024);    // 8 MB
  bf16_t* Vtb = (bf16_t*)(ws + (size_t)16 * 1024 * 1024);    // 8 MB
  bf16_t* Ctx = (bf16_t*)(ws + (size_t)24 * 1024 * 1024);    // 8 MB

  dim3 blk(128);
  dim3 gproj(DM / 64, M / 64);                  // (16, 64)

  hipLaunchKernelGGL((gemm_bias_kernel<0, false>), gproj, blk, 0, stream,
                     (const void*)queries, Wq, bq, (void*)Qbf, DM, DM);
  hipLaunchKernelGGL((gemm_bias_kernel<0, false>), gproj, blk, 0, stream,
                     (const void*)keys, Wk, bk, (void*)Kbf, DM, DM);
  hipLaunchKernelGGL((gemm_bias_kernel<1, false>), gproj, blk, 0, stream,
                     (const void*)values, Wv, bv, (void*)Vtb, DM, DM);

  attn_kernel<<<1024, blk, 0, stream>>>(Qbf, Kbf, Vtb, attnb, Ctx);

  hipLaunchKernelGGL((gemm_bias_kernel<2, true>), gproj, blk, 0, stream,
                     (const void*)Ctx, Wo, bo, d_out, DM, DM);
}
